// HyperTransformerPre_7739531067703
// MI455X (gfx1250) — matrix operand order repaired and checked by reading, not compiled, not run on hardware
//
#include <hip/hip_runtime.h>
#include <hip/hip_bf16.h>

// =====================================================================
// HyperTransformer pre-stage for MI455X (gfx1250, wave32, WMMA + TDM).
// Convs + channel-correlation GEMMs on v_wmma_f32_16x16x32_f16
// (f16 operands packed in registers/LDS, f32 accumulate).
// Channel-slice copies via the Tensor Data Mover (tensor_load_to_lds /
// tensor_store_from_lds + s_wait_tensorcnt).
// =====================================================================

typedef __attribute__((ext_vector_type(16))) _Float16     v16h;
typedef __attribute__((ext_vector_type(8)))  float        v8f;
typedef __attribute__((ext_vector_type(8)))  unsigned int v8u;
typedef __attribute__((ext_vector_type(4)))  unsigned int v4u;
typedef __attribute__((ext_vector_type(8)))  int          v8i;
typedef __attribute__((ext_vector_type(4)))  int          v4i;

#define MODE_NONE      0
#define MODE_RELU      1
#define MODE_ADD       2   // out = acc + bias + base
#define MODE_ADD_SCALE 3   // out = base + (acc + bias) * S[b,c]

#ifdef __gfx1250__
#if __has_builtin(__builtin_amdgcn_tensor_load_to_lds) && \
    __has_builtin(__builtin_amdgcn_tensor_store_from_lds) && \
    __has_builtin(__builtin_amdgcn_s_wait_tensorcnt)
#define HT_HAVE_TDM 1
#endif
#endif

// pack two f32 -> one dword of two f16 (RNE via scalar converts)
__device__ __forceinline__ unsigned int pack2h(float a, float b) {
  unsigned short lo = __builtin_bit_cast(unsigned short, (_Float16)a);
  unsigned short hi = __builtin_bit_cast(unsigned short, (_Float16)b);
  return (unsigned int)lo | ((unsigned int)hi << 16);
}

// =====================================================================
// Weight pre-pack: w[co][ci][kh][kw] f32 -> wp[co][kh][kw][ci] f16,
// zero-padded to CoutP x CinP so fragment loads are unconditional.
// =====================================================================
__global__ void pack_w_kernel(const float* __restrict__ w, _Float16* __restrict__ wp,
                              int Cout, int Cin, int ks, int CinP, int total)
{
  int idx = blockIdx.x * 256 + threadIdx.x;
  if (idx >= total) return;
  int ci = idx % CinP; int t = idx / CinP;
  int kw = t % ks; t /= ks;
  int kh = t % ks; int co = t / ks;
  float v = 0.f;
  if (co < Cout && ci < Cin)
    v = w[(((size_t)co * Cin + ci) * ks + kh) * ks + kw];
  wp[idx] = (_Float16)v;
}

// =====================================================================
// Implicit-GEMM conv: NCHW f32 in/out, f16 WMMA compute.
// Block = 128 threads (4 waves): 64 Cout x 16 pixels per block.
// K loop: ci chunks of 32 (f16-packed LDS slab), fully unrolled KSxKS taps.
// A fragment: 8 dword loads of packed f16 weights (global, L2-resident).
// B fragment: 8 dword LDS reads of packed f16 activations.
// =====================================================================
template <int KS>
__global__ __launch_bounds__(128) void conv_wmma_kernel(
    const float* __restrict__ x, const _Float16* __restrict__ wp,
    const float* __restrict__ bias, float* __restrict__ out,
    const float* __restrict__ base, const float* __restrict__ Sc,
    int Cin, int CinP, int Cout, int H, int W, int xtshift, int mode,
    int CoutBuf, int coutOff)
{
  constexpr int PAD = KS / 2;
  constexpr int CW  = 16 + KS - 1;
  __shared__ unsigned int slds[KS][CW][16];   // [row][col][ciPair] packed 2xf16

  const int tid    = threadIdx.x;
  const int lane   = tid & 31;
  const int wave   = tid >> 5;
  const int laneHi = lane >> 4;
  const int laneLo = lane & 15;
  const int pxBase = (blockIdx.x & ((1 << xtshift) - 1)) << 4;
  const int oy     = blockIdx.x >> xtshift;
  const int coutBase = blockIdx.y * 64 + wave * 16;
  const int b      = blockIdx.z;
  const int myCout = coutBase + laneLo;            // < CoutP (padded) always

  const float* xb = x + (size_t)b * Cin * H * W;
  const unsigned int* wq0 =
      (const unsigned int*)wp + (((size_t)myCout * KS * KS * CinP) >> 1);

  v8f acc = {};

  for (int ci0 = 0; ci0 < CinP; ci0 += 32) {
    // ---- stage packed-f16 input slab (zero-filled halo/channel tail) ----
    constexpr int NPAIR = KS * CW * 16;
    for (int e = tid; e < NPAIR; e += 128) {
      int pr = e & 15;
      int t2 = e >> 4;
      int c  = t2 % CW;
      int r  = t2 / CW;
      int gci = ci0 + pr * 2;
      int iy = oy + r - PAD, ix = pxBase + c - PAD;
      float f0 = 0.f, f1 = 0.f;
      if (iy >= 0 && iy < H && ix >= 0 && ix < W && gci < Cin) {
        size_t o0 = ((size_t)gci * H + iy) * W + ix;
        f0 = xb[o0];
        if (gci + 1 < Cin) f1 = xb[o0 + (size_t)H * W];
      }
      slds[r][c][pr] = pack2h(f0, f1);
    }
    __syncthreads();

    // prefetch next ci chunk (gfx1250 global_prefetch_b8)
    if (ci0 + 32 < Cin) {
      int pci = ci0 + 32 + lane; if (pci >= Cin) pci = Cin - 1;
      int py  = oy - PAD;        if (py < 0) py = 0;
      __builtin_prefetch(&xb[((size_t)pci * H + py) * W + pxBase], 0, 1);
    }

    // ---- KS*KS WMMAs over the staged chunk (fully unrolled) ----
#pragma unroll
    for (int kh = 0; kh < KS; ++kh) {
#pragma unroll
      for (int kw = 0; kw < KS; ++kw) {
        const unsigned int* wq = wq0 + ((((kh * KS) + kw) * CinP + ci0) >> 1);
        v8u au, bu;
#pragma unroll
        for (int v = 0; v < 8; ++v) {
          int pi = v + laneHi * 8;   // contiguous K map /2 (k = 16*laneHi + s)
          au[v] = wq[pi];
        }
#pragma unroll
        for (int v = 0; v < 8; ++v)
          bu[v] = slds[kh][laneLo + kw][v + laneHi * 8];     // contiguous K map /2 (same as A)
        acc = __builtin_amdgcn_wmma_f32_16x16x32_f16(
            false, __builtin_bit_cast(v16h, au),
            false, __builtin_bit_cast(v16h, bu),
            (short)0, acc, false, false);
      }
    }
    __syncthreads();
  }

  // ---- epilogue: bias + (relu | residual | residual*scale) ----
#pragma unroll
  for (int r = 0; r < 8; ++r) {
    int m    = r + laneHi * 8;
    int cout = coutBase + m;
    if (cout >= Cout) continue;
    int ox = pxBase + laneLo;
    float v = acc[r] + bias[cout];
    size_t bidx = (((size_t)b * Cout + cout) * H + oy) * W + ox;
    if (mode == MODE_RELU)           v = fmaxf(v, 0.f);
    else if (mode == MODE_ADD)       v += base[bidx];
    else if (mode == MODE_ADD_SCALE) v = base[bidx] + v * Sc[b * Cout + cout];
    out[(((size_t)b * CoutBuf + coutOff + cout) * H + oy) * W + ox] = v;
  }
}

// =====================================================================
// ts_hard: CC[b,i,j] = sum_n (K[b,i,n]-mK)(Q[b,j,n]-mQ) via WMMA.
// K-pairs are memory-contiguous -> float2 loads + packed f16 convert.
// 16-way HW split, f32 atomic reduction into CC.
// =====================================================================
__global__ __launch_bounds__(32) void cc_wmma_kernel(
    const float* __restrict__ kf, const float* __restrict__ qf,
    const float* __restrict__ mk, const float* __restrict__ mq,
    float* __restrict__ cc, int C, int HW)
{
  int lane   = threadIdx.x & 31;
  int laneHi = lane >> 4, laneLo = lane & 15;
  int itile  = blockIdx.x;
  int jtile  = blockIdx.y >> 4;     // NSPLIT = 16
  int split  = blockIdx.y & 15;
  int b      = blockIdx.z;
  int i = itile * 16 + laneLo;
  int j = jtile * 16 + laneLo;
  const float* kr = kf + ((size_t)b * C + i) * HW;
  const float* qr = qf + ((size_t)b * C + j) * HW;
  float mki = mk[b * C + i];
  float mqj = mq[b * C + j];
  int chunk = HW >> 4;
  int n0 = split * chunk, n1 = n0 + chunk;

  v8f acc = {};
  for (int n = n0; n < n1; n += 32) {
    v8u au, bu;
#pragma unroll
    for (int v = 0; v < 8; ++v) {
      int nA = n + 2 * v + laneHi * 16;
      float2 t = *(const float2*)(kr + nA);
      au[v] = pack2h(t.x - mki, t.y - mki);
    }
#pragma unroll
    for (int v = 0; v < 8; ++v) {
      int nB = n + 2 * v + laneHi * 16;
      float2 t = *(const float2*)(qr + nB);
      bu[v] = pack2h(t.x - mqj, t.y - mqj);
    }
    acc = __builtin_amdgcn_wmma_f32_16x16x32_f16(
        false, __builtin_bit_cast(v16h, au),
        false, __builtin_bit_cast(v16h, bu),
        (short)0, acc, false, false);
  }
#pragma unroll
  for (int r = 0; r < 8; ++r) {
    int mi = itile * 16 + r + laneHi * 8;
    int nj = jtile * 16 + laneLo;
    atomicAdd(&cc[((size_t)b * C + mi) * C + nj], acc[r]);
  }
}

// =====================================================================
// TDM channel-slice copy: 4096x8-f32 tiles through LDS.
// tensor_load_to_lds -> s_wait_tensorcnt -> tensor_store_from_lds.
// D# built per CDNA5 ISA 8.3/8.4 (2-D tensor, data_size=4B).
// =====================================================================
#ifdef HT_HAVE_TDM
__device__ __forceinline__ void tdm_desc(unsigned long long gaddr, unsigned int ldsAddr,
                                         unsigned int dim0, unsigned int dim1,
                                         unsigned int td0, unsigned int td1,
                                         unsigned long long stride0,
                                         v4u& g0, v8i& g1)
{
  g0[0] = 1u;                                        // count=1 (valid descriptor)
  g0[1] = ldsAddr;                                   // lds_addr (bytes)
  g0[2] = (unsigned int)(gaddr & 0xffffffffu);       // global_addr[31:0]
  g0[3] = (unsigned int)((gaddr >> 32) & 0x01ffffffu) | (2u << 30);  // [56:32]|type=2
  g1[0] = (int)(2u << 16);                                           // data_size=4B
  g1[1] = (int)((dim0 & 0xffffu) << 16);                             // tensor_dim0 lo
  g1[2] = (int)(((dim0 >> 16) & 0xffffu) | ((dim1 & 0xffffu) << 16));// dim0 hi|dim1 lo
  g1[3] = (int)(((dim1 >> 16) & 0xffffu) | ((td0 & 0xffffu) << 16)); // dim1 hi|tile0
  g1[4] = (int)(td1 & 0xffffu);                                      // tile_dim1
  g1[5] = (int)(stride0 & 0xffffffffu);                              // dim0_stride lo
  g1[6] = (int)((stride0 >> 32) & 0xffffu);                          // dim0_stride hi
  g1[7] = 0;
}
#endif

__global__ __launch_bounds__(32) void tdm_copy_kernel(
    const float* __restrict__ src, float* __restrict__ dst,
    int HW, int Csrc, int srcOff, int Cdst, int dstOff, int ctshift)
{
  const int TD0 = 4096, TD1 = 8;
  int b  = blockIdx.y;
  int ct = blockIdx.x & ((1 << ctshift) - 1);
  int rt = blockIdx.x >> ctshift;
  int c0 = ct * TD0;
  int r0 = rt * TD1;
  const float* sp = src + (((size_t)b * Csrc + srcOff + r0) * HW + c0);
  float*       dp = dst + (((size_t)b * Cdst + dstOff + r0) * HW + c0);
#ifdef HT_HAVE_TDM
  __shared__ float tile[TD0 * TD1];   // 128 KB staging tile at LDS offset 0
  (void)tile;
  v4u g0l, g0s; v8i g1l, g1s;
  tdm_desc((unsigned long long)(size_t)sp, 0u, TD0, TD1, TD0, TD1,
           (unsigned long long)(unsigned int)Csrc * (unsigned int)HW, g0l, g1l);
  tdm_desc((unsigned long long)(size_t)dp, 0u, TD0, TD1, TD0, TD1,
           (unsigned long long)(unsigned int)Cdst * (unsigned int)HW, g0s, g1s);
  v4i gz; gz[0] = 0; gz[1] = 0; gz[2] = 0; gz[3] = 0;
#if __clang_major__ >= 23
  v8i gz8; for (int q = 0; q < 8; ++q) gz8[q] = 0;
  __builtin_amdgcn_tensor_load_to_lds(g0l, g1l, gz, gz, gz8, 0);
  __builtin_amdgcn_s_wait_tensorcnt(0);
  __builtin_amdgcn_tensor_store_from_lds(g0s, g1s, gz, gz, gz8, 0);
  __builtin_amdgcn_s_wait_tensorcnt(0);
#else
  __builtin_amdgcn_tensor_load_to_lds(g0l, g1l, gz, gz, 0);
  __builtin_amdgcn_s_wait_tensorcnt(0);
  __builtin_amdgcn_tensor_store_from_lds(g0s, g1s, gz, gz, 0);
  __builtin_amdgcn_s_wait_tensorcnt(0);
#endif
#else
  for (int r = 0; r < TD1; ++r)
    for (int c = threadIdx.x; c < TD0; c += 32)
      dp[(size_t)r * HW + c] = sp[(size_t)r * HW + c];
#endif
}

// ---------------- mean over HW per (b,c) ----------------
__global__ __launch_bounds__(256) void mean_kernel(
    const float* __restrict__ xin, float* __restrict__ mout, int HW)
{
  __shared__ float red[256];
  size_t basei = (size_t)blockIdx.x * HW;
  float s = 0.f;
  for (int i = threadIdx.x; i < HW; i += 256) s += xin[basei + i];
  red[threadIdx.x] = s; __syncthreads();
  for (int off = 128; off > 0; off >>= 1) {
    if (threadIdx.x < off) red[threadIdx.x] += red[threadIdx.x + off];
    __syncthreads();
  }
  if (threadIdx.x == 0) mout[blockIdx.x] = red[0] / (float)HW;
}

// ---------------- column-wise max + argmax of CC ----------------
__global__ __launch_bounds__(32) void argmax_kernel(
    const float* __restrict__ cc, float* __restrict__ Sout,
    int* __restrict__ Hout, int C)
{
  int j = blockIdx.x, b = blockIdx.y;
  int lane = threadIdx.x;
  float best = -3.4e38f; int bidx = 0x7fffffff;
  for (int i = lane; i < C; i += 32) {
    float v = cc[((size_t)b * C + i) * C + j];
    if (v > best || (v == best && i < bidx)) { best = v; bidx = i; }
  }
  for (int m = 16; m > 0; m >>= 1) {
    float ov = __shfl_xor(best, m, 32);
    int   oi = __shfl_xor(bidx, m, 32);
    if (ov > best || (ov == best && oi < bidx)) { best = ov; bidx = oi; }
  }
  if (lane == 0) { Sout[b * C + j] = best; Hout[b * C + j] = bidx; }
}

// ---------------- gather V channels by argmax into concat slice ----------------
__global__ void gather_kernel(const float* __restrict__ V, const int* __restrict__ Hidx,
                              float* __restrict__ dst, int C, int HW,
                              int Cdst, int dstOff, int total)
{
  int idx = blockIdx.x * 256 + threadIdx.x;
  if (idx >= total) return;
  int p = idx % HW; int t = idx / HW;
  int c = t % C;    int b = t / C;
  int h = Hidx[b * C + c];
  dst[(((size_t)b * Cdst + dstOff + c) * HW) + p] = V[(((size_t)b * C + h) * HW) + p];
}

// ---------------- misc elementwise kernels ----------------
__global__ void zero_kernel(float* __restrict__ p, int n) {
  int i = blockIdx.x * 256 + threadIdx.x;
  if (i < n) p[i] = 0.f;
}

__global__ void maxpool_kernel(const float* __restrict__ in, float* __restrict__ out,
                               int C, int Hi, int Wi, int total)
{
  int idx = blockIdx.x * 256 + threadIdx.x;
  if (idx >= total) return;
  int Ho = Hi >> 1, Wo = Wi >> 1;
  int x = idx % Wo; int t = idx / Wo;
  int y = t % Ho;   t /= Ho;
  int c = t % C;    int b = t / C;
  const float* p = in + (((size_t)b * C + c) * Hi + 2 * y) * Wi + 2 * x;
  float v = fmaxf(fmaxf(p[0], p[1]), fmaxf(p[Wi], p[Wi + 1]));
  out[(((size_t)b * C + c) * Ho + y) * Wo + x] = v;
}

__global__ void pixshuf_kernel(const float* __restrict__ in, float* __restrict__ out,
                               int Cin, int Hi, int Wi, int relu, int total)
{
  int idx = blockIdx.x * 256 + threadIdx.x;
  if (idx >= total) return;
  int Co = Cin >> 2, Ho = Hi << 1, Wo = Wi << 1;
  int x = idx % Wo; int t = idx / Wo;
  int y = t % Ho;   t /= Ho;
  int c = t % Co;   int b = t / Co;
  int ci = c * 4 + (y & 1) * 2 + (x & 1);
  float v = in[(((size_t)b * Cin + ci) * Hi + (y >> 1)) * Wi + (x >> 1)];
  if (relu) v = fmaxf(v, 0.f);
  out[(((size_t)b * Co + c) * Ho + y) * Wo + x] = v;
}

__device__ __forceinline__ float cubw(float t) {   // Keys a = -0.5
  t = fabsf(t); float t2 = t * t, t3 = t2 * t;
  if (t <= 1.f) return 1.5f * t3 - 2.5f * t2 + 1.f;
  if (t <  2.f) return -0.5f * t3 + 2.5f * t2 - 4.f * t + 2.f;
  return 0.f;
}

__global__ void resize_bicubic_kernel(const float* __restrict__ in, float* __restrict__ out,
                                      int C, int Hi, int Wi, int Ho, int Wo,
                                      int Cdst, int dstOff, int total)
{
  int idx = blockIdx.x * 256 + threadIdx.x;
  if (idx >= total) return;
  int ox = idx % Wo; int t = idx / Wo;
  int oy = t % Ho;   t /= Ho;
  int c = t % C;     int b = t / C;
  float sy = (float)Hi / (float)Ho, sx = (float)Wi / (float)Wo;
  float fy = (oy + 0.5f) * sy - 0.5f;
  float fx = (ox + 0.5f) * sx - 0.5f;
  int iy = (int)floorf(fy), ix = (int)floorf(fx);
  const float* p = in + ((size_t)b * C + c) * Hi * Wi;
  float acc = 0.f;
#pragma unroll
  for (int dy = -1; dy <= 2; ++dy) {
    int yy = iy + dy; yy = yy < 0 ? 0 : (yy >= Hi ? Hi - 1 : yy);
    float wy = cubw(fy - (float)(iy + dy));
#pragma unroll
    for (int dx = -1; dx <= 2; ++dx) {
      int xx = ix + dx; xx = xx < 0 ? 0 : (xx >= Wi ? Wi - 1 : xx);
      acc += wy * cubw(fx - (float)(ix + dx)) * p[(size_t)yy * Wi + xx];
    }
  }
  out[(((size_t)b * Cdst + dstOff + c) * Ho + oy) * Wo + ox] = acc;
}

__global__ void resize_bilinear_kernel(const float* __restrict__ in, float* __restrict__ out,
                                       int C, int Hi, int Wi, int Ho, int Wo, int total)
{
  int idx = blockIdx.x * 256 + threadIdx.x;
  if (idx >= total) return;
  int ox = idx % Wo; int t = idx / Wo;
  int oy = t % Ho;   t /= Ho;
  int c = t % C;     int b = t / C;
  float sy = (float)Hi / (float)Ho, sx = (float)Wi / (float)Wo;
  float fy = (oy + 0.5f) * sy - 0.5f;
  float fx = (ox + 0.5f) * sx - 0.5f;
  int y0 = (int)floorf(fy), x0 = (int)floorf(fx);
  float ty = fy - (float)y0, tx = fx - (float)x0;
  int y0c = y0 < 0 ? 0 : (y0 >= Hi ? Hi - 1 : y0);
  int y1c = y0 + 1 < 0 ? 0 : (y0 + 1 >= Hi ? Hi - 1 : y0 + 1);
  int x0c = x0 < 0 ? 0 : (x0 >= Wi ? Wi - 1 : x0);
  int x1c = x0 + 1 < 0 ? 0 : (x0 + 1 >= Wi ? Wi - 1 : x0 + 1);
  const float* p = in + ((size_t)b * C + c) * Hi * Wi;
  float v00 = p[(size_t)y0c * Wi + x0c], v01 = p[(size_t)y0c * Wi + x1c];
  float v10 = p[(size_t)y1c * Wi + x0c], v11 = p[(size_t)y1c * Wi + x1c];
  float v = (1.f - ty) * ((1.f - tx) * v00 + tx * v01) + ty * ((1.f - tx) * v10 + tx * v11);
  out[(((size_t)b * C + c) * Ho + oy) * Wo + ox] = v;
}

// =====================================================================
// Host orchestration
// =====================================================================
static inline int ceildiv(int a, int b) { return (a + b - 1) / b; }

static void run_conv(hipStream_t s, const float* x, const float* w, const float* b,
                     float* out, const float* base, const float* Sc,
                     int Cin, int Cout, int H, int W, int ks, int mode,
                     int CoutBuf, int coutOff, _Float16* wpbuf)
{
  int CinP  = (Cin + 31) & ~31;
  int CoutP = (Cout + 63) & ~63;
  int total = CoutP * ks * ks * CinP;
  pack_w_kernel<<<ceildiv(total, 256), 256, 0, s>>>(w, wpbuf, Cout, Cin, ks, CinP, total);
  int xtshift = __builtin_ctz(W >> 4);
  dim3 grid((W / 16) * H, CoutP / 64, 2);
  if (ks == 7)
    conv_wmma_kernel<7><<<grid, 128, 0, s>>>(x, wpbuf, b, out, base, Sc,
                                             Cin, CinP, Cout, H, W, xtshift, mode,
                                             CoutBuf, coutOff);
  else
    conv_wmma_kernel<3><<<grid, 128, 0, s>>>(x, wpbuf, b, out, base, Sc,
                                             Cin, CinP, Cout, H, W, xtshift, mode,
                                             CoutBuf, coutOff);
}

static void run_copy(hipStream_t s, const float* src, float* dst,
                     int Cn, int HW, int Csrc, int srcOff, int Cdst, int dstOff)
{
  // TDM tile = 4096 x 8 f32; all call sites divide evenly.
  int colTiles = HW / 4096;
  int ctshift  = __builtin_ctz(colTiles);
  dim3 grid(colTiles * (Cn / 8), 2);
  tdm_copy_kernel<<<grid, 32, 0, s>>>(src, dst, HW, Csrc, srcOff, Cdst, dstOff, ctshift);
}

// jax pytree (sorted-dict) flat input order; each conv contributes (b, w)
enum {
  IN_XMS = 0, IN_XPAN = 1,
  P_C11H = 2, P_C12 = 4, P_C22H = 6, P_C23 = 8, P_C33H = 10, P_FIN = 12,
  P_HSI = 14,      // lfe: c128_1,c128_2,c256_1,c256_2,c64_1,c64_2 -> 12 slots
  P_PAN = 26,
  P_SFE_HEAD = 38, P_SFE_RB0 = 40 /* 16 x 4 */, P_SFE_TAIL = 104
};

extern "C" void kernel_launch(void* const* d_in, const int* in_sizes, int n_in,
                              void* d_out, int out_size, void* d_ws, size_t ws_size,
                              hipStream_t stream)
{
  (void)in_sizes; (void)n_in; (void)out_size; (void)ws_size;
  auto P = [&](int i) { return (const float*)d_in[i]; };
  const float* XMS  = P(IN_XMS);   // (2,102,64,64)
  const float* XPAN = P(IN_XPAN);  // (2,256,256) == (2,1,256,256)
  float* OUT = (float*)d_out;      // (2,102,256,256)

  float* wsf = (float*)d_ws;
  size_t off = 0;
  auto alloc = [&](size_t n) { float* p = wsf + off; off += n; return p; };

  const int B = 2;
  float* xmsup = alloc((size_t)B * 102 * 256 * 256);
  float* pand  = alloc((size_t)B * 64 * 64);
  float* panud = alloc((size_t)B * 256 * 256);
  float* V1 = alloc((size_t)B * 64 * 65536);
  float* V2 = alloc((size_t)B * 128 * 16384);
  float* V3 = alloc((size_t)B * 256 * 4096);
  float* K1 = alloc((size_t)B * 64 * 65536);
  float* K2 = alloc((size_t)B * 128 * 16384);
  float* K3 = alloc((size_t)B * 256 * 4096);
  float* Q1 = alloc((size_t)B * 64 * 65536);
  float* Q2 = alloc((size_t)B * 128 * 16384);
  float* Q3 = alloc((size_t)B * 256 * 4096);
  float* tA  = alloc((size_t)B * 64 * 65536);
  float* tP  = alloc((size_t)B * 64 * 16384);
  float* tB  = alloc((size_t)B * 128 * 16384);
  float* tP2 = alloc((size_t)B * 128 * 4096);
  float* tC  = alloc((size_t)B * 256 * 4096);
  float* mK = alloc(512);
  float* mQ = alloc(512);
  float* CC = alloc((size_t)B * 256 * 256);
  float* S3 = alloc(512); float* S2 = alloc(256); float* S1 = alloc(128);
  int* H3 = (int*)alloc(512); int* H2 = (int*)alloc(256); int* H1 = (int*)alloc(128);
  float* cat512  = alloc((size_t)B * 512 * 4096);
  float* cat256b = alloc((size_t)B * 256 * 16384);
  float* cat128c = alloc((size_t)B * 128 * 65536);
  float* xh   = alloc((size_t)B * 256 * 4096);   // sfe head output == x1
  float* xA   = alloc((size_t)B * 256 * 4096);
  float* xB   = alloc((size_t)B * 256 * 4096);
  float* trb  = alloc((size_t)B * 256 * 4096);
  float* x11  = alloc((size_t)B * 256 * 4096);
  float* x11n = alloc((size_t)B * 256 * 4096);
  float* c512 = alloc((size_t)B * 512 * 4096);
  float* x22  = alloc((size_t)B * 128 * 16384);
  float* x22n = alloc((size_t)B * 128 * 16384);
  float* c256c = alloc((size_t)B * 256 * 16384);
  float* x33  = alloc((size_t)B * 64 * 65536);
  float* x33n = alloc((size_t)B * 64 * 65536);
  float* xF   = alloc((size_t)B * 448 * 65536);
  // packed-f16 weight scratch (reused sequentially; max conv = 1,179,648 halves)
  _Float16* wpbuf = (_Float16*)alloc(1 << 20);   // 2M halves

  auto ew = [&](int total) { return dim3(ceildiv(total, 256)); };

  // ---------- stage 0: resizes ----------
  {
    int t = B * 102 * 256 * 256;
    resize_bicubic_kernel<<<ew(t), 256, 0, stream>>>(XMS, xmsup, 102, 64, 64, 256, 256, 102, 0, t);
    t = B * 64 * 64;
    resize_bilinear_kernel<<<ew(t), 256, 0, stream>>>(XPAN, pand, 1, 256, 256, 64, 64, t);
    t = B * 256 * 256;
    resize_bilinear_kernel<<<ew(t), 256, 0, stream>>>(pand, panud, 1, 64, 64, 256, 256, t);
  }

  // ---------- LFE pyramids ----------
  auto run_lfe = [&](int pb, const float* xin, int cin, float* o1, float* o2, float* o3) {
    run_conv(stream, xin, P(pb + 9), P(pb + 8), tA, nullptr, nullptr, cin, 64, 256, 256, 7, MODE_RELU, 64, 0, wpbuf);
    run_conv(stream, tA, P(pb + 11), P(pb + 10), o1, nullptr, nullptr, 64, 64, 256, 256, 3, MODE_RELU, 64, 0, wpbuf);
    int t = B * 64 * 16384;
    maxpool_kernel<<<ew(t), 256, 0, stream>>>(o1, tP, 64, 256, 256, t);
    run_conv(stream, tP, P(pb + 1), P(pb + 0), tB, nullptr, nullptr, 64, 128, 128, 128, 3, MODE_RELU, 128, 0, wpbuf);
    run_conv(stream, tB, P(pb + 3), P(pb + 2), o2, nullptr, nullptr, 128, 128, 128, 128, 3, MODE_RELU, 128, 0, wpbuf);
    t = B * 128 * 4096;
    maxpool_kernel<<<ew(t), 256, 0, stream>>>(o2, tP2, 128, 128, 128, t);
    run_conv(stream, tP2, P(pb + 5), P(pb + 4), tC, nullptr, nullptr, 128, 256, 64, 64, 3, MODE_RELU, 256, 0, wpbuf);
    run_conv(stream, tC, P(pb + 7), P(pb + 6), o3, nullptr, nullptr, 256, 256, 64, 64, 3, MODE_NONE, 256, 0, wpbuf);
  };
  run_lfe(P_PAN, XPAN,  1,   V1, V2, V3);
  run_lfe(P_PAN, panud, 1,   K1, K2, K3);
  run_lfe(P_HSI, xmsup, 102, Q1, Q2, Q3);

  // ---------- ts_hard at each level ----------
  auto run_ts = [&](const float* Vf, const float* Kf, const float* Qf, int C, int HW,
                    float* Sarr, int* Harr, float* Tdst, int Cdst, int dstOff) {
    mean_kernel<<<B * C, 256, 0, stream>>>(Kf, mK, HW);
    mean_kernel<<<B * C, 256, 0, stream>>>(Qf, mQ, HW);
    int nCC = B * C * C;
    zero_kernel<<<ew(nCC), 256, 0, stream>>>(CC, nCC);
    dim3 gcc(C / 16, (C / 16) * 16, B);
    cc_wmma_kernel<<<gcc, 32, 0, stream>>>(Kf, Qf, mK, mQ, CC, C, HW);
    argmax_kernel<<<dim3(C, B), 32, 0, stream>>>(CC, Sarr, Harr, C);
    int t = B * C * HW;
    gather_kernel<<<ew(t), 256, 0, stream>>>(Vf, Harr, Tdst, C, HW, Cdst, dstOff, t);
  };
  run_ts(V3, K3, Q3, 256, 4096,  S3, H3, cat512,  512, 256);
  run_ts(V2, K2, Q2, 128, 16384, S2, H2, cat256b, 256, 128);
  run_ts(V1, K1, Q1, 64,  65536, S1, H1, cat128c, 128, 64);

  // ---------- SFE trunk (head + 16 RBs + tail) ----------
  run_conv(stream, XMS, P(P_SFE_HEAD + 1), P(P_SFE_HEAD + 0), xh, nullptr, nullptr,
           102, 256, 64, 64, 3, MODE_RELU, 256, 0, wpbuf);
  float* cur = xh;
  for (int i = 0; i < 16; ++i) {
    int pb = P_SFE_RB0 + i * 4;   // c1_b, c1_w, c2_b, c2_w
    run_conv(stream, cur, P(pb + 1), P(pb + 0), trb, nullptr, nullptr,
             256, 256, 64, 64, 3, MODE_RELU, 256, 0, wpbuf);
    float* nxt = (i & 1) ? xB : xA;
    run_conv(stream, trb, P(pb + 3), P(pb + 2), nxt, cur, nullptr,
             256, 256, 64, 64, 3, MODE_ADD, 256, 0, wpbuf);
    cur = nxt;
  }
  run_conv(stream, cur, P(P_SFE_TAIL + 1), P(P_SFE_TAIL + 0), x11, xh, nullptr,
           256, 256, 64, 64, 3, MODE_ADD, 256, 0, wpbuf);

  // ---------- decoder ----------
  run_copy(stream, x11, cat512, 256, 4096, 256, 0, 512, 0);
  run_conv(stream, cat512, P(P_C11H + 1), P(P_C11H + 0), x11n, x11, S3,
           512, 256, 64, 64, 3, MODE_ADD_SCALE, 256, 0, wpbuf);
  run_conv(stream, x11n, P(P_C12 + 1), P(P_C12 + 0), c512, nullptr, nullptr,
           256, 512, 64, 64, 3, MODE_NONE, 512, 0, wpbuf);
  {
    int t = B * 128 * 16384;
    pixshuf_kernel<<<ew(t), 256, 0, stream>>>(c512, x22, 512, 64, 64, 1, t);
  }
  run_copy(stream, x22, cat256b, 128, 16384, 128, 0, 256, 0);
  run_conv(stream, cat256b, P(P_C22H + 1), P(P_C22H + 0), x22n, x22, S2,
           256, 128, 128, 128, 3, MODE_ADD_SCALE, 128, 0, wpbuf);
  run_conv(stream, x22n, P(P_C23 + 1), P(P_C23 + 0), c256c, nullptr, nullptr,
           128, 256, 128, 128, 3, MODE_NONE, 256, 0, wpbuf);
  {
    int t = B * 64 * 65536;
    pixshuf_kernel<<<ew(t), 256, 0, stream>>>(c256c, x33, 256, 128, 128, 1, t);
  }
  run_copy(stream, x33, cat128c, 64, 65536, 64, 0, 128, 0);
  run_conv(stream, cat128c, P(P_C33H + 1), P(P_C33H + 0), x33n, x33, S1,
           128, 64, 256, 256, 3, MODE_ADD_SCALE, 64, 0, wpbuf);

  // xF = concat(bicubic(x11n, x4), bicubic(x22n, x2), x33n)
  {
    int t = B * 256 * 65536;
    resize_bicubic_kernel<<<ew(t), 256, 0, stream>>>(x11n, xF, 256, 64, 64, 256, 256, 448, 0, t);
    t = B * 128 * 65536;
    resize_bicubic_kernel<<<ew(t), 256, 0, stream>>>(x22n, xF, 128, 128, 128, 256, 256, 448, 256, t);
  }
  run_copy(stream, x33n, xF, 64, 65536, 64, 0, 448, 384);
  run_conv(stream, xF, P(P_FIN + 1), P(P_FIN + 0), OUT, nullptr, nullptr,
           448, 102, 256, 256, 3, MODE_NONE, 102, 0, wpbuf);
}
